// EGNN_NET2_28613072126663
// MI455X (gfx1250) — matrix operand order repaired and checked by reading, not compiled, not run on hardware
//
#include <hip/hip_runtime.h>
#include <hip/hip_bf16.h>
#include <math.h>

// EGNN forward for MI455X (gfx1250): bf16 WMMA for every dense GEMM,
// fused edge pipeline, weights repacked once per launch into bf16
// transposed [N][Kpad] layout so B fragments are two b128 loads per lane.

#define L_LAYERS 4
#define HIDF 128
#define EMBF 64
#define INDIM 20
#define SSF 3
#define EINK 321
#define EIN_PAD 352   // padded to multiple of 32 for WMMA K
#define NGRAPH 64

typedef __bf16 bf16_t;
typedef bf16_t v16bf __attribute__((ext_vector_type(16)));
typedef bf16_t v8bf  __attribute__((ext_vector_type(8)));
typedef float  v8f   __attribute__((ext_vector_type(8)));

__device__ __forceinline__ float siluf(float x) { return x / (1.0f + __expf(-x)); }
__device__ __forceinline__ float geluf(float x) {
  float x3 = x * x * x;
  return 0.5f * x * (1.0f + tanhf(0.7978845608028654f * (x + 0.044715f * x3)));
}

// ---------- WMMA helpers (wave32, v_wmma_f32_16x16x32_bf16) ----------
// A: 16(M) x kpad(K) bf16 tile in LDS, row stride kpad (16B-aligned rows).
// B: bf16 weights repacked transposed [N][kpad] in global (zero-padded K).
// One 16x16 D tile: lane L -> n = n0+(L&15), m rows r+8*(L>>4).
__device__ __forceinline__ v8f gemm_tile16(const bf16_t* __restrict__ As,
                                           const bf16_t* __restrict__ Wt,
                                           int kpad, int n0, int lane) {
  v8f acc = {};
  const bf16_t* ap = As + (lane & 15) * kpad + ((lane >> 4) << 3);
  const bf16_t* bp = Wt + (size_t)(n0 + (lane & 15)) * kpad + ((lane >> 4) << 3);
  for (int kc = 0; kc < kpad; kc += 32) {
    v8bf a0 = *(const v8bf*)(ap + kc);
    v8bf a1 = *(const v8bf*)(ap + kc + 16);
    v8bf b0 = *(const v8bf*)(bp + kc);
    v8bf b1 = *(const v8bf*)(bp + kc + 16);
    v16bf a, b;
#pragma unroll
    for (int j = 0; j < 8; ++j) {
      a[j] = a0[j]; a[8 + j] = a1[j];
      b[j] = b0[j]; b[8 + j] = b1[j];
    }
    acc = __builtin_amdgcn_wmma_f32_16x16x32_bf16(false, a, false, b, (short)0, acc,
                                                  false, false);
  }
  return acc;
}

// ---------- weight repack: fp32 [K][N] -> bf16 [N][Kpad], zero-padded ----------
__global__ void k_wconv(const float* __restrict__ W, int K, int N, int Kpad,
                        bf16_t* __restrict__ out) {
  size_t total = (size_t)N * Kpad;
  for (size_t i = (size_t)blockIdx.x * blockDim.x + threadIdx.x; i < total;
       i += (size_t)gridDim.x * blockDim.x) {
    int n = (int)(i / Kpad), k = (int)(i % Kpad);
    out[i] = (bf16_t)((k < K) ? W[(size_t)k * N + n] : 0.0f);
  }
}

// ---------- utility kernels ----------
__global__ void k_zero(float* p, size_t n) {
  for (size_t i = (size_t)blockIdx.x * blockDim.x + threadIdx.x; i < n;
       i += (size_t)gridDim.x * blockDim.x)
    p[i] = 0.0f;
}

__global__ void k_counts(int E, int Nn, const int* __restrict__ eidx,
                         const int* __restrict__ batch, float* cnt, float* gcnt) {
  int total = E + Nn;
  for (int i = blockIdx.x * blockDim.x + threadIdx.x; i < total;
       i += gridDim.x * blockDim.x) {
    if (i < E) atomicAdd(&cnt[eidx[E + i]], 1.0f);
    else       atomicAdd(&gcnt[batch[i - E]], 1.0f);
  }
}

__global__ void k_eattr(int E, const float* __restrict__ attr,
                        const float* __restrict__ eeW, const float* __restrict__ eeb,
                        float* __restrict__ e_attr) {
  size_t total = (size_t)E * 64;
  for (size_t i = (size_t)blockIdx.x * blockDim.x + threadIdx.x; i < total;
       i += (size_t)gridDim.x * blockDim.x) {
    size_t e = i >> 6; int j = (int)(i & 63);
    e_attr[i] = attr[e] * eeW[j] + eeb[j];
  }
}

// ---------- node embedding init ----------
__global__ void __launch_bounds__(256)
k_node_init(int Nn, const float* __restrict__ x, const float* __restrict__ b_pos,
            const float* __restrict__ b_type,
            const float* ssW1, const float* ssb1, const float* ssW2, const float* ssb2,
            const float* beW1, const float* beb1, const float* beW2, const float* beb2,
            float* __restrict__ feats, float* __restrict__ coors) {
  __shared__ float sh[256];
  int v = blockIdx.x; if (v >= Nn) return;
  int tid = threadIdx.x;
  if (tid < 128) {
    float h = ssb1[tid];
    for (int k = 0; k < INDIM; ++k) h += x[(size_t)v * INDIM + k] * ssW1[k * 128 + tid];
    sh[tid] = siluf(h);
  } else {
    int j = tid - 128;
    float h = beb1[j];
    for (int k = 0; k < SSF; ++k) h += b_type[(size_t)v * SSF + k] * beW1[k * 128 + j];
    sh[tid] = siluf(h);
  }
  __syncthreads();
  if (tid < 64) {
    float o = ssb2[tid];
    for (int k = 0; k < 128; ++k) o += sh[k] * ssW2[k * 64 + tid];
    feats[(size_t)v * HIDF + 64 + tid] = o;          // x_embed -> cols 64..127
  } else if (tid < 128) {
    int j = tid - 64;
    float o = beb2[j];
    for (int k = 0; k < 128; ++k) o += sh[128 + k] * beW2[k * 64 + j];
    feats[(size_t)v * HIDF + j] = o;                 // b_embed -> cols 0..63
  } else if (tid < 131) {
    int d = tid - 128;
    coors[(size_t)v * 3 + d] = b_pos[(size_t)v * 3 + d];
  }
}

// ---------- time embedding + FiLM tables ----------
__global__ void __launch_bounds__(128)
k_time(int T, const float* __restrict__ timev,
       const float* tW1, const float* tb1, const float* tW2, const float* tb2,
       const float* Wt, const float* bt, float* __restrict__ te_all) {
  __shared__ float s_sin[128], s_h[128], s_ts[64];
  int t = blockIdx.x; if (t >= T) return;
  int tid = threadIdx.x;
  float tv = timev[t];
  if (tid < 64) {
    float f = __expf((float)tid * (-logf(10000.0f) / 63.0f));
    float a = tv * f;
    s_sin[tid] = sinf(a);
    s_sin[64 + tid] = cosf(a);
  }
  __syncthreads();
  {
    float h = tb1[tid];
    for (int k = 0; k < 128; ++k) h += s_sin[k] * tW1[k * 128 + tid];
    s_h[tid] = siluf(h);
  }
  __syncthreads();
  if (tid < 64) {
    float v = tb2[tid];
    for (int k = 0; k < 128; ++k) v += s_h[k] * tW2[k * 64 + tid];
    s_ts[tid] = siluf(v);          // only silu(t) is consumed downstream
  }
  __syncthreads();
  for (int l = 0; l < L_LAYERS; ++l) {
#pragma unroll
    for (int half = 0; half < 2; ++half) {
      int n = tid + half * 128;
      float v = bt[l * 256 + n];
      for (int k = 0; k < 64; ++k) v += s_ts[k] * Wt[((size_t)l * 64 + k) * 256 + n];
      te_all[((size_t)l * 64 + t) * 256 + n] = v;
    }
  }
}

// ---------- fused edge pipeline (the heavy kernel) ----------
// Per block: 16 edges, 4 waves; waves split the N dimension of each GEMM.
__global__ void __launch_bounds__(128)
k_edge(int E, const int* __restrict__ eidx,
       const float* __restrict__ coors, const float* __restrict__ feats,
       float* __restrict__ e_attr, float* __restrict__ m_i, float* __restrict__ coor_acc,
       const bf16_t* We1t, const float* be1, const bf16_t* We2t, const float* be2,
       const bf16_t* Wut,  const float* bu,
       const bf16_t* Wc1t, const float* bc1, const float* Wc2, const float* bc2) {
  __shared__ __align__(16) bf16_t s_ein[16 * EIN_PAD];
  __shared__ __align__(16) bf16_t s_h1[16 * 128];
  __shared__ __align__(16) bf16_t s_m [16 * 128];
  __shared__ float  s_mf[16 * 128];
  __shared__ __align__(16) bf16_t s_h2[16 * 512];
  __shared__ float  s_rel[16][4];
  __shared__ float  s_w[16];
  __shared__ float  s_red[128];
  __shared__ int    s_src[16], s_dst[16];

  const int tid  = threadIdx.x;
  const int lane = tid & 31;
  const int wave = tid >> 5;
  const int e0   = blockIdx.x * 16;

  if (tid < 16) {
    int e = e0 + tid; if (e >= E) e = E - 1;
    int s = eidx[e], d = eidx[E + e];
    s_src[tid] = s; s_dst[tid] = d;
    float rx = coors[(size_t)s * 3 + 0] - coors[(size_t)d * 3 + 0];
    float ry = coors[(size_t)s * 3 + 1] - coors[(size_t)d * 3 + 1];
    float rz = coors[(size_t)s * 3 + 2] - coors[(size_t)d * 3 + 2];
    s_rel[tid][0] = rx; s_rel[tid][1] = ry; s_rel[tid][2] = rz;
    s_rel[tid][3] = rx * rx + ry * ry + rz * rz;
    // rel_d column + zero pad tail
    s_ein[tid * EIN_PAD + 256] = (bf16_t)s_rel[tid][3];
    for (int k = EINK; k < EIN_PAD; ++k) s_ein[tid * EIN_PAD + k] = (bf16_t)0.0f;
  }
  __syncthreads();

  // e_in = [feats[dst] | feats[src] | rel_d | e_attr]  (float4-vectorized gather)
  for (int idx = tid; idx < 16 * 64; idx += 128) {
    int m = idx >> 6, q = idx & 63;
    const float* sp = (q < 32) ? (feats + (size_t)s_dst[m] * HIDF + q * 4)
                               : (feats + (size_t)s_src[m] * HIDF + (q - 32) * 4);
    float4 f = *(const float4*)sp;
    bf16_t* o = s_ein + m * EIN_PAD + q * 4;
    o[0] = (bf16_t)f.x; o[1] = (bf16_t)f.y; o[2] = (bf16_t)f.z; o[3] = (bf16_t)f.w;
  }
  for (int idx = tid; idx < 16 * 16; idx += 128) {
    int m = idx >> 4, q = idx & 15;
    int e = e0 + m; if (e >= E) e = E - 1;
    float4 f = *(const float4*)(e_attr + (size_t)e * 64 + q * 4);
    bf16_t* o = s_ein + m * EIN_PAD + 257 + q * 4;
    o[0] = (bf16_t)f.x; o[1] = (bf16_t)f.y; o[2] = (bf16_t)f.z; o[3] = (bf16_t)f.w;
  }
  __syncthreads();

  // GEMM1: (16 x 321) @ We1(321 x 128) -> silu -> s_h1
  {
    int nb = wave * 32;
#pragma unroll
    for (int t = 0; t < 2; ++t) {
      int n0 = nb + t * 16;
      v8f acc = gemm_tile16(s_ein, We1t, EIN_PAD, n0, lane);
      int n = n0 + (lane & 15), mb = (lane >> 4) << 3;
#pragma unroll
      for (int r = 0; r < 8; ++r)
        s_h1[(mb + r) * 128 + n] = (bf16_t)siluf(acc[r] + be1[n]);
    }
  }
  __syncthreads();

  // GEMM2: (16 x 128) @ We2(128 x 128) -> silu -> m
  {
    int nb = wave * 32;
#pragma unroll
    for (int t = 0; t < 2; ++t) {
      int n0 = nb + t * 16;
      v8f acc = gemm_tile16(s_h1, We2t, 128, n0, lane);
      int n = n0 + (lane & 15), mb = (lane >> 4) << 3;
#pragma unroll
      for (int r = 0; r < 8; ++r) {
        float mv = siluf(acc[r] + be2[n]);
        s_m [(mb + r) * 128 + n] = (bf16_t)mv;
        s_mf[(mb + r) * 128 + n] = mv;
      }
    }
  }
  __syncthreads();

  // e_attr update: m @ Wu(128 x 64) -> silu -> global (in place, per-edge rows)
  if (wave < 2) {
    int nb = wave * 32;
#pragma unroll
    for (int t = 0; t < 2; ++t) {
      int n0 = nb + t * 16;
      v8f acc = gemm_tile16(s_m, Wut, 128, n0, lane);
      int n = n0 + (lane & 15), mb = (lane >> 4) << 3;
#pragma unroll
      for (int r = 0; r < 8; ++r) {
        int e = e0 + mb + r;
        if (e < E) e_attr[(size_t)e * 64 + n] = siluf(acc[r] + bu[n]);
      }
    }
  }

  // coord MLP hidden: m @ Wc1(128 x 512) -> silu -> s_h2
  for (int t = 0; t < 8; ++t) {
    int n0 = wave * 128 + t * 16;
    v8f acc = gemm_tile16(s_m, Wc1t, 128, n0, lane);
    int n = n0 + (lane & 15), mb = (lane >> 4) << 3;
#pragma unroll
    for (int r = 0; r < 8; ++r)
      s_h2[(mb + r) * 512 + n] = (bf16_t)siluf(acc[r] + bc1[n]);
  }
  __syncthreads();

  // Wc2(512 x 1): per-edge dot product, 8 threads per edge
  {
    int e = tid >> 3, part = tid & 7;
    const bf16_t* hp = s_h2 + e * 512 + part * 64;
    const float*  wp = Wc2 + part * 64;
    float s = 0.0f;
    for (int k = 0; k < 64; ++k) s += (float)hp[k] * wp[k];
    s_red[tid] = s;
  }
  __syncthreads();
  if (tid < 16) {
    float s = 0.0f;
#pragma unroll
    for (int j = 0; j < 8; ++j) s += s_red[tid * 8 + j];
    s_w[tid] = s + bc2[0];
  }
  __syncthreads();

  // scatter: coor_acc[dst] += rel * w ; m_i[dst] += m
  if (tid < 48) {
    int m = tid / 3, d = tid - m * 3;
    int e = e0 + m;
    if (e < E) atomicAdd(&coor_acc[(size_t)s_dst[m] * 3 + d], s_rel[m][d] * s_w[m]);
  }
  for (int idx = tid; idx < 16 * 128; idx += 128) {
    int m = idx >> 7;
    int e = e0 + m;
    if (e < E) atomicAdd(&m_i[(size_t)s_dst[m] * HIDF + (idx & 127)], s_mf[idx]);
  }
}

// ---------- node update: coors += acc/cnt ; feats += MLP([feats|m_i]) ; FiLM ----------
__global__ void __launch_bounds__(128)
k_node_update(int Nn, const float* __restrict__ m_i, const float* __restrict__ coor_acc,
              const float* __restrict__ cnt, float* __restrict__ coors,
              float* __restrict__ feats, const int* __restrict__ batch,
              const float* __restrict__ te,
              const bf16_t* Wn1t, const float* bn1, const bf16_t* Wn2t, const float* bn2) {
  __shared__ __align__(16) bf16_t s_a[16 * 256];
  __shared__ __align__(16) bf16_t s_h[16 * 128];
  int tid = threadIdx.x, lane = tid & 31, wave = tid >> 5;
  int base = blockIdx.x * 16;

  if (tid < 48) {
    int m = tid / 3, d = tid - m * 3;
    int v = base + m;
    if (v < Nn) {
      float c = cnt[v]; if (c < 1.0f) c = 1.0f;
      coors[(size_t)v * 3 + d] += coor_acc[(size_t)v * 3 + d] / c;
    }
  }
  for (int idx = tid; idx < 16 * 64; idx += 128) {
    int m = idx >> 6, q = idx & 63;
    int v = base + m; if (v >= Nn) v = Nn - 1;
    const float* sp = (q < 32) ? (feats + (size_t)v * HIDF + q * 4)
                               : (m_i + (size_t)v * HIDF + (q - 32) * 4);
    float4 f = *(const float4*)sp;
    bf16_t* o = s_a + m * 256 + q * 4;
    o[0] = (bf16_t)f.x; o[1] = (bf16_t)f.y; o[2] = (bf16_t)f.z; o[3] = (bf16_t)f.w;
  }
  __syncthreads();

  {
    int nb = wave * 32;
#pragma unroll
    for (int t = 0; t < 2; ++t) {
      int n0 = nb + t * 16;
      v8f acc = gemm_tile16(s_a, Wn1t, 256, n0, lane);
      int n = n0 + (lane & 15), mb = (lane >> 4) << 3;
#pragma unroll
      for (int r = 0; r < 8; ++r)
        s_h[(mb + r) * 128 + n] = (bf16_t)siluf(acc[r] + bn1[n]);
    }
  }
  __syncthreads();

  {
    int nb = wave * 32;
#pragma unroll
    for (int t = 0; t < 2; ++t) {
      int n0 = nb + t * 16;
      v8f acc = gemm_tile16(s_h, Wn2t, 128, n0, lane);
      int n = n0 + (lane & 15), mb = (lane >> 4) << 3;
#pragma unroll
      for (int r = 0; r < 8; ++r) {
        int v = base + mb + r;
        if (v < Nn) {
          int g = batch[v];
          float sc = te[(size_t)g * 256 + n];
          float sf = te[(size_t)g * 256 + 128 + n];
          float f = feats[(size_t)v * HIDF + n] + acc[r] + bn2[n];
          feats[(size_t)v * HIDF + n] = f * (sc + 1.0f) + sf;
        }
      }
    }
  }
}

// ---------- graph layernorm stats ----------
__global__ void __launch_bounds__(128)
k_stats(int Nn, const float* __restrict__ feats, const int* __restrict__ batch,
        float* gsum, float* gsq) {
  __shared__ float r1[128], r2[128];
  int v = blockIdx.x; if (v >= Nn) return;
  int tid = threadIdx.x;
  float x = feats[(size_t)v * HIDF + tid];
  r1[tid] = x; r2[tid] = x * x;
  __syncthreads();
  for (int s = 64; s > 0; s >>= 1) {
    if (tid < s) { r1[tid] += r1[tid + s]; r2[tid] += r2[tid + s]; }
    __syncthreads();
  }
  if (tid == 0) {
    int g = batch[v];
    atomicAdd(&gsum[g], r1[0]);
    atomicAdd(&gsq[g],  r2[0]);
  }
}

__global__ void k_ln_fin(const float* gcnt, const float* gsum, const float* gsq,
                         float* gmean, float* grstd) {
  int g = threadIdx.x;
  if (g < NGRAPH) {
    float denom = gcnt[g] * (float)HIDF; if (denom < 1.0f) denom = 1.0f;
    float mean = gsum[g] / denom;
    float var = gsq[g] / denom - mean * mean;
    if (var < 0.0f) var = 0.0f;
    gmean[g] = mean;
    grstd[g] = rsqrtf(var + 1e-5f);
  }
}

// ---------- FFN: fn = LN(feats); feats = W2(gelu(W1 fn)) + fn ----------
__global__ void __launch_bounds__(128)
k_ffn(int Nn, float* __restrict__ feats, const int* __restrict__ batch,
      const float* __restrict__ gmean, const float* __restrict__ grstd,
      const float* __restrict__ gamma, const float* __restrict__ beta,
      const bf16_t* Wf1t, const float* bf1, const bf16_t* Wf2t, const float* bf2) {
  __shared__ __align__(16) bf16_t s_fn[16 * 128];
  __shared__ float  s_fnf[16 * 128];
  __shared__ __align__(16) bf16_t s_h[16 * 512];
  int tid = threadIdx.x, lane = tid & 31, wave = tid >> 5;
  int base = blockIdx.x * 16;

  for (int idx = tid; idx < 16 * 32; idx += 128) {
    int m = idx >> 5, q = idx & 31;
    int v = base + m; if (v >= Nn) v = Nn - 1;
    int g = batch[v];
    float mean = gmean[g], rs = grstd[g];
    float4 f = *(const float4*)(feats + (size_t)v * HIDF + q * 4);
    int n = q * 4;
    float a0 = (f.x - mean) * rs * gamma[n + 0] + beta[n + 0];
    float a1 = (f.y - mean) * rs * gamma[n + 1] + beta[n + 1];
    float a2 = (f.z - mean) * rs * gamma[n + 2] + beta[n + 2];
    float a3 = (f.w - mean) * rs * gamma[n + 3] + beta[n + 3];
    float* of = s_fnf + m * 128 + n;
    of[0] = a0; of[1] = a1; of[2] = a2; of[3] = a3;
    bf16_t* ob = s_fn + m * 128 + n;
    ob[0] = (bf16_t)a0; ob[1] = (bf16_t)a1; ob[2] = (bf16_t)a2; ob[3] = (bf16_t)a3;
  }
  __syncthreads();

  for (int t = 0; t < 8; ++t) {
    int n0 = wave * 128 + t * 16;
    v8f acc = gemm_tile16(s_fn, Wf1t, 128, n0, lane);
    int n = n0 + (lane & 15), mb = (lane >> 4) << 3;
#pragma unroll
    for (int r = 0; r < 8; ++r)
      s_h[(mb + r) * 512 + n] = (bf16_t)geluf(acc[r] + bf1[n]);
  }
  __syncthreads();

  {
    int nb = wave * 32;
#pragma unroll
    for (int t = 0; t < 2; ++t) {
      int n0 = nb + t * 16;
      v8f acc = gemm_tile16(s_h, Wf2t, 512, n0, lane);
      int n = n0 + (lane & 15), mb = (lane >> 4) << 3;
#pragma unroll
      for (int r = 0; r < 8; ++r) {
        int v = base + mb + r;
        if (v < Nn)
          feats[(size_t)v * HIDF + n] = acc[r] + bf2[n] + s_fnf[(mb + r) * 128 + n];
      }
    }
  }
}

// ---------- output head ----------
__global__ void __launch_bounds__(128)
k_out(int Nn, const float* __restrict__ feats,
      const float* dW1, const float* db1, const float* dW2, const float* db2,
      float* __restrict__ out) {
  __shared__ float s_f[128], s_h[128];
  int v = blockIdx.x; if (v >= Nn) return;
  int tid = threadIdx.x;
  s_f[tid] = feats[(size_t)v * HIDF + tid];
  __syncthreads();
  float h = db1[tid];
  for (int k = 0; k < 128; ++k) h += s_f[k] * dW1[k * 128 + tid];
  s_h[tid] = siluf(h);
  __syncthreads();
  if (tid < INDIM + SSF) {
    float o = db2[tid];
    for (int k = 0; k < 128; ++k) o += s_h[k] * dW2[k * (INDIM + SSF) + tid];
    if (tid < INDIM) out[(size_t)v * INDIM + tid] = o;
    else out[(size_t)Nn * INDIM + (size_t)v * SSF + (tid - INDIM)] = o;
  }
}

extern "C" void kernel_launch(void* const* d_in, const int* in_sizes, int n_in,
                              void* d_out, int out_size, void* d_ws, size_t ws_size,
                              hipStream_t stream) {
  const float* x      = (const float*)d_in[0];
  const float* b_pos  = (const float*)d_in[1];
  const int*   eidx   = (const int*)  d_in[2];
  const float* battr  = (const float*)d_in[3];
  const float* b_type = (const float*)d_in[4];
  const int*   batch  = (const int*)  d_in[5];
  const float* timev  = (const float*)d_in[6];
  const float* ssW1 = (const float*)d_in[7];  const float* ssb1 = (const float*)d_in[8];
  const float* ssW2 = (const float*)d_in[9];  const float* ssb2 = (const float*)d_in[10];
  const float* tW1  = (const float*)d_in[11]; const float* tb1  = (const float*)d_in[12];
  const float* tW2  = (const float*)d_in[13]; const float* tb2  = (const float*)d_in[14];
  const float* beW1 = (const float*)d_in[15]; const float* beb1 = (const float*)d_in[16];
  const float* beW2 = (const float*)d_in[17]; const float* beb2 = (const float*)d_in[18];
  const float* eeW  = (const float*)d_in[19]; const float* eeb  = (const float*)d_in[20];
  const float* We1  = (const float*)d_in[21]; const float* be1  = (const float*)d_in[22];
  const float* We2  = (const float*)d_in[23]; const float* be2  = (const float*)d_in[24];
  const float* Wu   = (const float*)d_in[25]; const float* bu   = (const float*)d_in[26];
  const float* Wc1  = (const float*)d_in[27]; const float* bc1  = (const float*)d_in[28];
  const float* Wc2  = (const float*)d_in[29]; const float* bc2  = (const float*)d_in[30];
  const float* Wn1  = (const float*)d_in[31]; const float* bn1  = (const float*)d_in[32];
  const float* Wn2  = (const float*)d_in[33]; const float* bn2  = (const float*)d_in[34];
  const float* Wt   = (const float*)d_in[35]; const float* bt   = (const float*)d_in[36];
  const float* gam  = (const float*)d_in[37]; const float* bet  = (const float*)d_in[38];
  const float* Wf1  = (const float*)d_in[39]; const float* bf1  = (const float*)d_in[40];
  const float* Wf2  = (const float*)d_in[41]; const float* bf2  = (const float*)d_in[42];
  const float* dW1  = (const float*)d_in[43]; const float* db1  = (const float*)d_in[44];
  const float* dW2  = (const float*)d_in[45]; const float* db2  = (const float*)d_in[46];
  (void)n_in; (void)out_size; (void)ws_size;

  const int Nn = in_sizes[0] / INDIM;     // 20000
  const int E  = in_sizes[3];             // b_edge_attr is (E,1) -> 320000
  const int T  = in_sizes[6];             // 64 timesteps / graphs

  // ---- workspace layout ----
  float* ws = (float*)d_ws;
  size_t off = 0;
  float* feats = ws + off;  off += (size_t)Nn * HIDF;
  float* coors = ws + off;  off += (size_t)Nn * 3;
  float* e_attr = ws + off; off += (size_t)E * 64;
  float* cntz  = ws + off;  off += (size_t)Nn + NGRAPH;   // cnt_dst | gcnt
  float* cnt   = cntz;
  float* gcnt  = cntz + Nn;
  float* gmean = ws + off;  off += NGRAPH;
  float* grstd = ws + off;  off += NGRAPH;
  float* te_all = ws + off; off += (size_t)L_LAYERS * 64 * 256;
  float* zl    = ws + off;  off += (size_t)Nn * HIDF + (size_t)Nn * 3 + 2 * NGRAPH;
  float* m_i      = zl;
  float* coor_acc = zl + (size_t)Nn * HIDF;
  float* gsum     = coor_acc + (size_t)Nn * 3;
  float* gsq      = gsum + NGRAPH;
  const size_t zl_len = (size_t)Nn * HIDF + (size_t)Nn * 3 + 2 * NGRAPH;

  // bf16 repacked weights (16B-aligned region)
  off = (off + 7) & ~(size_t)7;
  bf16_t* wb = (bf16_t*)(ws + off);
  size_t boff = 0;
  auto ba = [&](size_t n) { bf16_t* p = wb + boff; boff += n; return p; };
  const size_t we1s = 128 * (size_t)EIN_PAD, we2s = 128 * 128, wus = 64 * 128,
               wc1s = 512 * 128, wn1s = 128 * 256, wn2s = 128 * 128,
               wf1s = 512 * 128, wf2s = 128 * 512;
  bf16_t* We1t = ba(L_LAYERS * we1s);
  bf16_t* We2t = ba(L_LAYERS * we2s);
  bf16_t* Wut  = ba(L_LAYERS * wus);
  bf16_t* Wc1t = ba(L_LAYERS * wc1s);
  bf16_t* Wn1t = ba(L_LAYERS * wn1s);
  bf16_t* Wn2t = ba(L_LAYERS * wn2s);
  bf16_t* Wf1t = ba(L_LAYERS * wf1s);
  bf16_t* Wf2t = ba(L_LAYERS * wf2s);

  const int etiles = (E + 15) / 16;
  const int ntiles = (Nn + 15) / 16;

  // weight repack (cheap; L2-resident afterwards)
  for (int l = 0; l < L_LAYERS; ++l) {
    k_wconv<<<64, 256, 0, stream>>>(We1 + (size_t)l * EINK * 128, EINK, 128, EIN_PAD,
                                    We1t + l * we1s);
    k_wconv<<<32, 256, 0, stream>>>(We2 + (size_t)l * 128 * 128, 128, 128, 128,
                                    We2t + l * we2s);
    k_wconv<<<16, 256, 0, stream>>>(Wu + (size_t)l * 128 * 64, 128, 64, 128,
                                    Wut + l * wus);
    k_wconv<<<64, 256, 0, stream>>>(Wc1 + (size_t)l * 128 * 512, 128, 512, 128,
                                    Wc1t + l * wc1s);
    k_wconv<<<32, 256, 0, stream>>>(Wn1 + (size_t)l * 256 * 128, 256, 128, 256,
                                    Wn1t + l * wn1s);
    k_wconv<<<16, 256, 0, stream>>>(Wn2 + (size_t)l * 128 * 128, 128, 128, 128,
                                    Wn2t + l * wn2s);
    k_wconv<<<64, 256, 0, stream>>>(Wf1 + (size_t)l * 128 * 512, 128, 512, 128,
                                    Wf1t + l * wf1s);
    k_wconv<<<64, 256, 0, stream>>>(Wf2 + (size_t)l * 512 * 128, 512, 128, 512,
                                    Wf2t + l * wf2s);
  }

  k_zero<<<256, 256, 0, stream>>>(cntz, (size_t)Nn + NGRAPH);
  k_counts<<<512, 256, 0, stream>>>(E, Nn, eidx, batch, cnt, gcnt);
  k_node_init<<<Nn, 256, 0, stream>>>(Nn, x, b_pos, b_type, ssW1, ssb1, ssW2, ssb2,
                                      beW1, beb1, beW2, beb2, feats, coors);
  k_time<<<T, 128, 0, stream>>>(T, timev, tW1, tb1, tW2, tb2, Wt, bt, te_all);
  k_eattr<<<2048, 256, 0, stream>>>(E, battr, eeW, eeb, e_attr);

  for (int l = 0; l < L_LAYERS; ++l) {
    k_zero<<<1024, 256, 0, stream>>>(zl, zl_len);
    k_edge<<<etiles, 128, 0, stream>>>(E, eidx, coors, feats, e_attr, m_i, coor_acc,
        We1t + l * we1s, be1 + l * 128,
        We2t + l * we2s, be2 + l * 128,
        Wut  + l * wus,  bu  + l * 64,
        Wc1t + l * wc1s, bc1 + l * 512,
        Wc2 + (size_t)l * 512, bc2 + l);
    k_node_update<<<ntiles, 128, 0, stream>>>(Nn, m_i, coor_acc, cnt, coors, feats,
        batch, te_all + (size_t)l * 64 * 256,
        Wn1t + l * wn1s, bn1 + l * 128,
        Wn2t + l * wn2s, bn2 + l * 128);
    k_stats<<<Nn, 128, 0, stream>>>(Nn, feats, batch, gsum, gsq);
    k_ln_fin<<<1, 64, 0, stream>>>(gcnt, gsum, gsq, gmean, grstd);
    k_ffn<<<ntiles, 128, 0, stream>>>(Nn, feats, batch, gmean, grstd,
        gam + l * 128, bet + l * 128,
        Wf1t + l * wf1s, bf1 + l * 512,
        Wf2t + l * wf2s, bf2 + l * 128);
  }
  k_out<<<Nn, 128, 0, stream>>>(Nn, feats, dW1, db1, dW2, db2, (float*)d_out);
}
